// Decoder_43705587204650
// MI455X (gfx1250) — compile-verified, matrix operand order checked by reading, not run on hardware
//
#include <hip/hip_runtime.h>
#include <hip/hip_bf16.h>
#include <math.h>
#include <stdint.h>

// MI455X / gfx1250 implementation.
// Phase 1 (parallel): enc_atts = enc @ We^T + be via bf16 WMMA (16x16x32), f32 accum.
//   Inputs pre-converted to bf16; tiles staged global->LDS by the Tensor Data Mover
//   (tensor_load_to_lds) with TDM padding to a 40-element LDS row stride,
//   double-buffered on TENSORcnt.
// Phase 2 (sequential scan over 256 steps): persistent 16-workgroup kernel,
//   global software barrier, per-step [16x512]x[512x512] WMMA per WGP,
//   TDM prefetch of the step's enc_atts rows overlapped with the dec_att GEMV,
//   all recurrent weights cached as bf16 (L2-resident on a 192MB-L2 part).

typedef unsigned short ushort_t;
typedef unsigned int   uint_t;
typedef __attribute__((ext_vector_type(16))) __bf16  v16bf;
typedef __attribute__((ext_vector_type(8)))  float   v8f;
typedef __attribute__((ext_vector_type(8)))  uint_t  v8u;
typedef __attribute__((ext_vector_type(4)))  unsigned int tdm_v4u;
typedef __attribute__((ext_vector_type(8)))  int  tdm_v8i;
typedef __attribute__((ext_vector_type(4)))  int  tdm_v4i;

#define NWG 16
#define ROWS_PER_WG 13   // 16 * 13 = 208 >= 196

#if __has_builtin(__builtin_amdgcn_tensor_load_to_lds)
#define HAVE_TDM 1
#else
#define HAVE_TDM 0
#endif

__device__ __forceinline__ ushort_t f2bf(float x){
  uint_t u = __builtin_bit_cast(uint_t, x);
  uint_t r = u + 0x7FFFu + ((u >> 16) & 1u);   // round-to-nearest-even
  return (ushort_t)(r >> 16);
}
__device__ __forceinline__ float bf2f(ushort_t s){
  return __builtin_bit_cast(float, ((uint_t)s) << 16);
}
__device__ __forceinline__ float sigmoidf_(float x){ return 1.0f/(1.0f + __expf(-x)); }

// Load one 16x32 bf16 fragment (A-layout per CDNA5 ISA 7.12.2):
// lane = m (0..15) | hi*16 ; VGPR v holds K pair kb(v,hi) = v<4 ? 2v+8*hi : 8+2v+8*hi.
// The two 4-VGPR halves are contiguous 16B chunks -> compiler emits ds_load_b128 pairs.
__device__ __forceinline__ v16bf load_frag(const ushort_t* base, int hi){
  v8u tt;
#pragma unroll
  for (int v = 0; v < 8; ++v){
    int kb = (v < 4) ? (2*v + 8*hi) : (8 + 2*v + 8*hi);
    tt[v] = *(const uint_t*)(base + kb);        // kb even -> dword aligned
  }
  return __builtin_bit_cast(v16bf, tt);
}

// 2D Tensor-DMA: global -> LDS. Descriptor per CDNA5 ISA 8.3/8.4.
// data_size_code: 0=1B,1=2B,2=4B. Dims/strides in data_size units.
// pad_*_code per ISA: interval 0=2,1=4,2=8,3=16 dwords; amount 0=1..3=4 dwords.
__device__ __forceinline__ void tdm_load_2d(
    const void* gsrc, void* ldst,
    unsigned data_size_code,
    unsigned tile_d0, unsigned tile_d1,
    unsigned tensor_d0, unsigned tensor_d1,
    unsigned stride0_elems,
    unsigned pad_interval_code, unsigned pad_amount_code, unsigned pad_en)
{
#if HAVE_TDM
  unsigned long long ga = (unsigned long long)(uintptr_t)gsrc;
  unsigned int la = (unsigned int)(uintptr_t)ldst;   // flat LDS addr: low 32 bits = LDS offset
  tdm_v4u g0;
  g0[0] = 1u;                                        // count=1, user descriptor
  g0[1] = la;                                        // lds_addr (bytes)
  g0[2] = (unsigned int)(ga & 0xFFFFFFFFull);        // global_addr[31:0]
  g0[3] = (unsigned int)((ga >> 32) & 0x01FFFFFFull) | (2u << 30);  // addr[56:32] | type=2
  tdm_v8i g1;
  g1[0] = (int)((data_size_code << 16) | (pad_en << 20) |
                (pad_interval_code << 22) | (pad_amount_code << 25));
  g1[1] = (int)((tensor_d0 & 0xFFFFu) << 16);                            // dim0[15:0]
  g1[2] = (int)(((tensor_d0 >> 16) & 0xFFFFu) | ((tensor_d1 & 0xFFFFu) << 16));
  g1[3] = (int)(((tensor_d1 >> 16) & 0xFFFFu) | ((tile_d0 & 0xFFFFu) << 16));
  g1[4] = (int)(tile_d1 & 0xFFFFu);                                      // tile_dim1, tile_dim2=0
  g1[5] = (int)stride0_elems;                                            // dim0_stride[31:0]
  g1[6] = 0;
  g1[7] = 0;
  tdm_v4i gz; gz[0] = 0; gz[1] = 0; gz[2] = 0; gz[3] = 0;
#if __clang_major__ >= 23
  __builtin_amdgcn_tensor_load_to_lds(g0, g1, gz, gz, g1, 0);
#else
  __builtin_amdgcn_tensor_load_to_lds(g0, g1, gz, gz, 0);
#endif
#else
  (void)gsrc; (void)ldst; (void)data_size_code; (void)tile_d0; (void)tile_d1;
  (void)tensor_d0; (void)tensor_d1; (void)stride0_elems;
  (void)pad_interval_code; (void)pad_amount_code; (void)pad_en;
#endif
}
__device__ __forceinline__ void tdm_wait(){
#if HAVE_TDM
#if __has_builtin(__builtin_amdgcn_s_wait_tensorcnt)
  __builtin_amdgcn_s_wait_tensorcnt(0);
#endif
#endif
}

__device__ __forceinline__ void global_barrier(int* cnt, int* gen, int nwg){
  __syncthreads();
  __threadfence();
  if (threadIdx.x == 0){
    int g = __hip_atomic_load(gen, __ATOMIC_ACQUIRE, __HIP_MEMORY_SCOPE_AGENT);
    int prev = __hip_atomic_fetch_add(cnt, 1, __ATOMIC_ACQ_REL, __HIP_MEMORY_SCOPE_AGENT);
    if (prev == nwg - 1){
      __hip_atomic_store(cnt, 0, __ATOMIC_RELAXED, __HIP_MEMORY_SCOPE_AGENT);
      __hip_atomic_fetch_add(gen, 1, __ATOMIC_ACQ_REL, __HIP_MEMORY_SCOPE_AGENT);
    } else {
      while (__hip_atomic_load(gen, __ATOMIC_ACQUIRE, __HIP_MEMORY_SCOPE_AGENT) == g)
        __builtin_amdgcn_s_sleep(2);
    }
  }
  __syncthreads();
  __threadfence();
}

// ---------------------------------------------------------------- prep kernels
__global__ __launch_bounds__(256) void prep_main(
    const float* __restrict__ enc,  const float* __restrict__ We,
    const float* __restrict__ W_ih, const float* __restrict__ W_hh,
    const float* __restrict__ Wd,   const float* __restrict__ Wf1,
    const float* __restrict__ bn1_g, const float* __restrict__ bn1_b,
    const float* __restrict__ bn1_m, const float* __restrict__ bn1_v,
    const float* __restrict__ bn2_g, const float* __restrict__ bn2_v,
    const float* __restrict__ Wf2,
    ushort_t* __restrict__ enc_bf, ushort_t* __restrict__ we_bf,
    ushort_t* __restrict__ wih_bf, ushort_t* __restrict__ whh_bf,
    ushort_t* __restrict__ wd_bf,  ushort_t* __restrict__ wf1_bf,
    float* __restrict__ bn1s, float* __restrict__ bn1b, float* __restrict__ wf2eff,
    float* __restrict__ h_g, float* __restrict__ c_g, int* __restrict__ bar)
{
  size_t gid = (size_t)blockIdx.x * blockDim.x + threadIdx.x;
  size_t stride = (size_t)gridDim.x * blockDim.x;
  for (size_t i = gid; i < (size_t)50176*2048; i += stride) enc_bf[i] = f2bf(enc[i]);
  for (size_t i = gid; i < (size_t)512*2048;  i += stride) we_bf[i]  = f2bf(We[i]);
  for (size_t i = gid; i < (size_t)2048*2051; i += stride) wih_bf[i] = f2bf(W_ih[i]);
  for (size_t i = gid; i < (size_t)2048*512;  i += stride) whh_bf[i] = f2bf(W_hh[i]);
  for (size_t i = gid; i < (size_t)512*512;   i += stride){
    wd_bf[i]  = f2bf(Wd[i]);
    wf1_bf[i] = f2bf(Wf1[i]);
  }
  for (size_t i = gid; i < 512; i += stride){
    float s1 = bn1_g[i] * rsqrtf(bn1_v[i] + 1e-5f);
    bn1s[i] = s1;
    bn1b[i] = bn1_b[i] - bn1_m[i]*s1;
    float s2 = bn2_g[i] * rsqrtf(bn2_v[i] + 1e-5f);
    wf2eff[i] = s2 * Wf2[i];                     // bn2 scale folded into Wf2
    h_g[i] = 0.f; c_g[i] = 0.f;
  }
  if (gid < 2) bar[gid] = 0;
}

__global__ __launch_bounds__(256) void prep_c2(
    const float* __restrict__ bn2_g, const float* __restrict__ bn2_b,
    const float* __restrict__ bn2_m, const float* __restrict__ bn2_v,
    const float* __restrict__ Wf2,   const float* __restrict__ bf2,
    float* __restrict__ c2)
{
  __shared__ float red[256];
  int t = threadIdx.x;
  float p = 0.f;
  for (int a = t; a < 512; a += 256){
    float s2 = bn2_g[a] * rsqrtf(bn2_v[a] + 1e-5f);
    p += (bn2_b[a] - bn2_m[a]*s2) * Wf2[a];      // bn2 shift folded into a scalar
  }
  red[t] = p; __syncthreads();
  for (int s = 128; s > 0; s >>= 1){ if (t < s) red[t] += red[t+s]; __syncthreads(); }
  if (t == 0) c2[0] = red[0] + bf2[0];
}

// ------------------------------------------------- phase 1: enc @ We^T + be
// M = 50176, N = 512, K = 2048. Block tile 128x128, 8 waves, wave = 16(M) x 128(N).
// bf16 inputs; tiles staged by TDM (pad to 40-elem LDS rows), double-buffered on
// TENSORcnt. Fallback: manual bf16 copies when the TDM builtin is unavailable.
__global__ __launch_bounds__(256) void encatt_gemm(
    const ushort_t* __restrict__ Abf,  // [50176, 2048] bf16
    const ushort_t* __restrict__ Wbf,  // [512, 2048]   bf16
    const float* __restrict__ bias,    // [512]
    float* __restrict__ out)           // [50176, 512]
{
  __shared__ __align__(16) ushort_t a_lds[2][128*40];
  __shared__ __align__(16) ushort_t b_lds[2][128*40];
  const int t  = threadIdx.x;
  const int M0 = blockIdx.y * 128;
  const int N0 = blockIdx.x * 128;
  const int wv = t >> 5, lane = t & 31, m16 = lane & 15, hi = lane >> 4;

  v8f acc[8];
#pragma unroll
  for (int j = 0; j < 8; ++j)
#pragma unroll
    for (int r = 0; r < 8; ++r) acc[j][r] = 0.f;

#if HAVE_TDM
  if (wv == 0){
    tdm_load_2d(Abf + (size_t)M0*2048, &a_lds[0][0], 1, 32, 128, 32, 128, 2048, 3, 3, 1);
    tdm_load_2d(Wbf + (size_t)N0*2048, &b_lds[0][0], 1, 32, 128, 32, 128, 2048, 3, 3, 1);
    tdm_wait();
  }
  __syncthreads();
  int p = 0;
  for (int k0 = 0; k0 < 2048; k0 += 32){
    const bool has_next = (k0 + 32) < 2048;
    if (has_next && wv == 0){
      tdm_load_2d(Abf + (size_t)M0*2048 + k0 + 32, &a_lds[p ^ 1][0],
                  1, 32, 128, 32, 128, 2048, 3, 3, 1);
      tdm_load_2d(Wbf + (size_t)N0*2048 + k0 + 32, &b_lds[p ^ 1][0],
                  1, 32, 128, 32, 128, 2048, 3, 3, 1);
    }
    v16bf af = load_frag(&a_lds[p][(wv*16 + m16)*40], hi);
#pragma unroll
    for (int j = 0; j < 8; ++j){
      v16bf bfr = load_frag(&b_lds[p][(j*16 + m16)*40], hi);
      acc[j] = __builtin_amdgcn_wmma_f32_16x16x32_bf16(
          false, af, false, bfr, (short)0, acc[j], false, false);
    }
    if (has_next && wv == 0) tdm_wait();
    __syncthreads();
    if (has_next) p ^= 1;
  }
#else
  // manual-copy double buffering fallback
  for (int i = t; i < 128*32; i += 256){
    int m = i >> 5, k = i & 31;
    a_lds[0][m*40 + k] = Abf[(size_t)(M0 + m)*2048 + k];
    b_lds[0][m*40 + k] = Wbf[(size_t)(N0 + m)*2048 + k];
  }
  __syncthreads();
  int p = 0;
  for (int k0 = 0; k0 < 2048; k0 += 32){
    const bool has_next = (k0 + 32) < 2048;
    ushort_t av[16], bv[16];
    if (has_next){
#pragma unroll
      for (int i = 0; i < 16; ++i){
        int m = (t >> 5) + 8*i, k = t & 31;
        av[i] = Abf[(size_t)(M0 + m)*2048 + k0 + 32 + k];
        bv[i] = Wbf[(size_t)(N0 + m)*2048 + k0 + 32 + k];
      }
    }
    v16bf af = load_frag(&a_lds[p][(wv*16 + m16)*40], hi);
#pragma unroll
    for (int j = 0; j < 8; ++j){
      v16bf bfr = load_frag(&b_lds[p][(j*16 + m16)*40], hi);
      acc[j] = __builtin_amdgcn_wmma_f32_16x16x32_bf16(
          false, af, false, bfr, (short)0, acc[j], false, false);
    }
    if (has_next){
#pragma unroll
      for (int i = 0; i < 16; ++i){
        int m = (t >> 5) + 8*i, k = t & 31;
        a_lds[p ^ 1][m*40 + k] = av[i];
        b_lds[p ^ 1][m*40 + k] = bv[i];
      }
      p ^= 1;
    }
    __syncthreads();
  }
#endif

#pragma unroll
  for (int j = 0; j < 8; ++j){
    int col = N0 + j*16 + m16;
    float bi = bias[col];
#pragma unroll
    for (int r = 0; r < 8; ++r){
      int row = M0 + wv*16 + r + 8*hi;     // C layout: VGPR r -> M = r + 8*hi
      out[(size_t)row*512 + col] = acc[j][r] + bi;
    }
  }
}

// ------------------------------------------------- phase 2: persistent scan
__global__ __launch_bounds__(256) void scan_kernel(
    const float* __restrict__ enc,        // [256,196,2048]
    const float* __restrict__ enc_atts,   // [50176,512]
    const float* __restrict__ actions,    // [256,3]
    const ushort_t* __restrict__ wd_bf, const float* __restrict__ bd,
    const float* __restrict__ bn1s, const float* __restrict__ bn1b,
    const ushort_t* __restrict__ wf1_bf, const float* __restrict__ bf1,
    const float* __restrict__ wf2eff, const float* __restrict__ c2p,
    const ushort_t* __restrict__ wih_bf, const float* __restrict__ W_ih_f32,
    const ushort_t* __restrict__ whh_bf,
    const float* __restrict__ b_ih, const float* __restrict__ b_hh,
    const float* __restrict__ Wo1, const float* __restrict__ bo1,
    const float* __restrict__ Wo2, const float* __restrict__ bo2,
    float* __restrict__ h_g, float* __restrict__ c_g,
    float* __restrict__ score_g, float* __restrict__ att_g,
    int* bar, float* __restrict__ out)
{
  __shared__ float h_lds[512];
  __shared__ float dec_lds[512];
  __shared__ __align__(16) ushort_t x_lds[16*520];
  __shared__ float sc_lds[16];
  __shared__ float alpha_lds[208];
  __shared__ float red[256];
  __shared__ float gates_lds[128];
  // 32KB staging buffer for this step's enc_atts rows (TDM dest), unioned with
  // the later-stage att/z buffers (disjoint lifetimes, barrier-separated).
  __shared__ __align__(16) union LateU {
    float enc_raw[16*512];
    struct { float att[2048]; float z[512]; } late;
  } u;

  const int t = threadIdx.x;
  const int w = blockIdx.x;
  const int wv = t >> 5, lane = t & 31, m16 = lane & 15, hi = lane >> 4;
  const int lbase = w * ROWS_PER_WG;
  const float c2 = c2p[0];

  for (int b = 0; b < 256; ++b){
#if HAVE_TDM
    // async DMA of this WG's enc_atts rows, overlapped with the dec_att GEMV
    if (wv == 0){
      unsigned remaining = (unsigned)(50176 - (b*196 + lbase));
      tdm_load_2d(enc_atts + ((size_t)b*196 + lbase)*512, u.enc_raw,
                  2, 512, ROWS_PER_WG, 512, remaining, 512, 0, 0, 0);
    }
#endif
    // ---- A: stage h; dec_att = Wd @ h + bd (redundant per WG, Wd bf16 in L2)
    for (int i = t; i < 512; i += 256) h_lds[i] = h_g[i];
    if (t < 16) sc_lds[t] = 0.f;
    __syncthreads();
    for (int a = t; a < 512; a += 256){
      float s = bd[a];
      const ushort_t* wr = wd_bf + (size_t)a*512;
      for (int k = 0; k < 512; ++k) s += bf2f(wr[k]) * h_lds[k];
      dec_lds[a] = s;
    }
#if HAVE_TDM
    if (wv == 0) tdm_wait();
#endif
    __syncthreads();
    // ---- B: x = bn1(relu(enc_att + dec_att)) -> bf16 tile [16(pad) x 512]
    for (int i = t; i < 16*512; i += 256){
      int r = i >> 9, a = i & 511;
      int l = lbase + r;
      float v = 0.f;
      if (r < ROWS_PER_WG && l < 196){
#if HAVE_TDM
        float e = u.enc_raw[r*512 + a] + dec_lds[a];
#else
        float e = enc_atts[((size_t)b*196 + l)*512 + a] + dec_lds[a];
#endif
        e = fmaxf(e, 0.f);
        v = e*bn1s[a] + bn1b[a];
      }
      x_lds[r*520 + a] = f2bf(v);
    }
    __syncthreads();
    // ---- C: y1 = x @ Wf1^T via WMMA: each wave does 4 N-tiles of 16
    v8f acc[4];
#pragma unroll
    for (int j = 0; j < 4; ++j)
#pragma unroll
      for (int r = 0; r < 8; ++r) acc[j][r] = 0.f;
    for (int kk = 0; kk < 512; kk += 32){
      v16bf af = load_frag(&x_lds[m16*520 + kk], hi);
#pragma unroll
      for (int j = 0; j < 4; ++j){
        int n = (wv*4 + j)*16 + m16;                 // B[k][n] = Wf1[n][k]
        v16bf bfr = load_frag(&wf1_bf[(size_t)n*512 + kk], hi);
        acc[j] = __builtin_amdgcn_wmma_f32_16x16x32_bf16(
            false, af, false, bfr, (short)0, acc[j], false, false);
      }
    }
    // ---- D: score[l] = sum_n relu(y1 + bf1)[n] * wf2eff[n]  (+ c2 later)
    {
      float part[8];
#pragma unroll
      for (int r = 0; r < 8; ++r) part[r] = 0.f;
#pragma unroll
      for (int j = 0; j < 4; ++j){
        int n = (wv*4 + j)*16 + m16;
        float b1 = bf1[n], we = wf2eff[n];
#pragma unroll
        for (int r = 0; r < 8; ++r)
          part[r] += fmaxf(acc[j][r] + b1, 0.f) * we;
      }
#pragma unroll
      for (int r = 0; r < 8; ++r) atomicAdd(&sc_lds[r + 8*hi], part[r]);  // ds_add_f32
    }
    __syncthreads();
    if (t < ROWS_PER_WG && lbase + t < 196) score_g[lbase + t] = sc_lds[t] + c2;
    global_barrier(bar, bar + 1, NWG);
    // ---- E: softmax over 196 (redundant per WG)
    float sv = (t < 196) ? score_g[t] : -1e30f;
    red[t] = sv; __syncthreads();
    for (int s = 128; s > 0; s >>= 1){ if (t < s) red[t] = fmaxf(red[t], red[t+s]); __syncthreads(); }
    float smax = red[0]; __syncthreads();
    float ev = (t < 196) ? __expf(sv - smax) : 0.f;
    red[t] = ev; __syncthreads();
    for (int s = 128; s > 0; s >>= 1){ if (t < s) red[t] += red[t+s]; __syncthreads(); }
    float inv = 1.f / red[0]; __syncthreads();
    if (t < 196) alpha_lds[t] = ev * inv;
    __syncthreads();
    // ---- F: att_out chunk (128 cols per WG): alpha @ enc[b]
    {
      int col = t >> 1, half = t & 1, e0 = w*128;
      float s = 0.f;
      for (int l = half; l < 196; l += 2)
        s += alpha_lds[l] * enc[((size_t)b*196 + l)*2048 + e0 + col];
      red[t] = s; __syncthreads();
      if (half == 0) att_g[e0 + col] = red[t] + red[t+1];
    }
    global_barrier(bar, bar + 1, NWG);
    // ---- G: gates for d-chunk [w*32, w*32+32): all 4 gate rows, bf16 weights
    for (int i = t; i < 2048; i += 256) u.late.att[i] = att_g[i];
    __syncthreads();
    {
      int idx = t >> 1, half = t & 1;
      int grp = idx >> 5, dloc = idx & 31;
      int row = w*32 + dloc + grp*512;               // torch gate order i,f,g,o
      const ushort_t* wi = wih_bf + (size_t)row*2051;
      float s = 0.f;
      int kb = half*1024;
      for (int k = kb; k < kb + 1024; ++k) s += bf2f(wi[k]) * u.late.att[k];
      const ushort_t* wh = whh_bf + (size_t)row*512;
      int hb = half*256;
      for (int k = hb; k < hb + 256; ++k) s += bf2f(wh[k]) * h_lds[k];
      if (half == 0){
        const float* wif = W_ih_f32 + (size_t)row*2051 + 2048;
        for (int j = 0; j < 3; ++j) s += wif[j] * actions[b*3 + j];
        s += b_ih[row] + b_hh[row];
      }
      red[t] = s;
    }
    __syncthreads();
    if ((t & 1) == 0) gates_lds[t >> 1] = red[t] + red[t+1];
    __syncthreads();
    // ---- H: LSTM cell update for this WG's 32 hidden dims
    if (t < 32){
      int dd = w*32 + t;
      float i_ = sigmoidf_(gates_lds[t]);
      float f_ = sigmoidf_(gates_lds[32 + t]);
      float g_ = tanhf(gates_lds[64 + t]);
      float o_ = sigmoidf_(gates_lds[96 + t]);
      float cn = f_ * c_g[dd] + i_ * g_;
      c_g[dd] = cn;
      h_g[dd] = o_ * tanhf(cn);
    }
    global_barrier(bar, bar + 1, NWG);
    // ---- I: output MLP (WG0 only; overlaps with others starting next step)
    if (w == 0){
      for (int i = t; i < 512; i += 256) h_lds[i] = h_g[i];
      __syncthreads();
      for (int a = t; a < 512; a += 256){
        float s = bo1[a];
        const float* r1 = Wo1 + (size_t)a*512;
        for (int k = 0; k < 512; ++k) s += r1[k] * h_lds[k];
        u.late.z[a] = fmaxf(s, 0.f);
      }
      __syncthreads();
      if (t < 3){
        float s = bo2[t];
        const float* r2 = Wo2 + (size_t)t*512;
        for (int a = 0; a < 512; ++a) s += r2[a] * u.late.z[a];
        out[b*3 + t] = sigmoidf_(s);
      }
      __syncthreads();
    }
  }
}

// ---------------------------------------------------------------- launch
extern "C" void kernel_launch(void* const* d_in, const int* in_sizes, int n_in,
                              void* d_out, int out_size, void* d_ws, size_t ws_size,
                              hipStream_t stream) {
  const float* enc     = (const float*)d_in[0];
  const float* actions = (const float*)d_in[1];
  const float* We      = (const float*)d_in[2];
  const float* be      = (const float*)d_in[3];
  const float* Wd      = (const float*)d_in[4];
  const float* bd      = (const float*)d_in[5];
  const float* bn1_g   = (const float*)d_in[6];
  const float* bn1_b   = (const float*)d_in[7];
  const float* bn1_m   = (const float*)d_in[8];
  const float* bn1_v   = (const float*)d_in[9];
  const float* Wf1     = (const float*)d_in[10];
  const float* bf1     = (const float*)d_in[11];
  const float* bn2_g   = (const float*)d_in[12];
  const float* bn2_b   = (const float*)d_in[13];
  const float* bn2_m   = (const float*)d_in[14];
  const float* bn2_v   = (const float*)d_in[15];
  const float* Wf2     = (const float*)d_in[16];
  const float* bf2     = (const float*)d_in[17];
  const float* W_ih    = (const float*)d_in[18];
  const float* W_hh    = (const float*)d_in[19];
  const float* b_ih    = (const float*)d_in[20];
  const float* b_hh    = (const float*)d_in[21];
  const float* Wo1     = (const float*)d_in[22];
  const float* bo1     = (const float*)d_in[23];
  const float* Wo2     = (const float*)d_in[24];
  const float* bo2     = (const float*)d_in[25];
  float* out = (float*)d_out;

  char* ws = (char*)d_ws;
  size_t off = 0;
  auto carve = [&](size_t bytes)->char* {
    char* p = ws + off;
    off += (bytes + 255) & ~(size_t)255;
    return p;
  };
  float*    enc_atts = (float*)   carve((size_t)50176*512*4);   // 102.8 MB
  ushort_t* enc_bf   = (ushort_t*)carve((size_t)50176*2048*2);  // 205.5 MB
  ushort_t* we_bf    = (ushort_t*)carve((size_t)512*2048*2);    //   2.1 MB
  ushort_t* wih_bf   = (ushort_t*)carve((size_t)2048*2051*2);   //   8.4 MB
  ushort_t* whh_bf   = (ushort_t*)carve((size_t)2048*512*2);
  ushort_t* wd_bf    = (ushort_t*)carve((size_t)512*512*2);
  ushort_t* wf1_bf   = (ushort_t*)carve((size_t)512*512*2);
  float*    bn1s     = (float*)   carve(512*4);
  float*    bn1b     = (float*)   carve(512*4);
  float*    wf2eff   = (float*)   carve(512*4);
  float*    c2       = (float*)   carve(256);
  float*    h_g      = (float*)   carve(512*4);
  float*    c_g      = (float*)   carve(512*4);
  float*    score_g  = (float*)   carve(256*4);
  float*    att_g    = (float*)   carve(2048*4);
  int*      bar      = (int*)     carve(256);
  (void)ws_size; (void)in_sizes; (void)n_in; (void)out_size;

  prep_main<<<dim3(4096), dim3(256), 0, stream>>>(
      enc, We, W_ih, W_hh, Wd, Wf1, bn1_g, bn1_b, bn1_m, bn1_v, bn2_g, bn2_v, Wf2,
      enc_bf, we_bf, wih_bf, whh_bf, wd_bf, wf1_bf, bn1s, bn1b, wf2eff, h_g, c_g, bar);
  prep_c2<<<dim3(1), dim3(256), 0, stream>>>(bn2_g, bn2_b, bn2_m, bn2_v, Wf2, bf2, c2);
  encatt_gemm<<<dim3(4, 392), dim3(256), 0, stream>>>(enc_bf, we_bf, be, enc_atts);
  scan_kernel<<<dim3(NWG), dim3(256), 0, stream>>>(
      enc, enc_atts, actions,
      wd_bf, bd, bn1s, bn1b, wf1_bf, bf1, wf2eff, c2,
      wih_bf, W_ih, whh_bf, b_ih, b_hh, Wo1, bo1, Wo2, bo2,
      h_g, c_g, score_g, att_g, bar, out);
}
